// Seq2Latent_30348238914190
// MI455X (gfx1250) — matrix operand order repaired and checked by reading, not compiled, not run on hardware
//
#include <hip/hip_runtime.h>

// ---------------- problem constants ----------------
constexpr int BATCH = 4;
constexpr int HH    = 48;
constexpr int WW    = 48;
constexpr int TT    = 16;
constexpr int KCH   = 64;                 // hidden channels per layer
constexpr int HWSZ  = HH * WW;            // 2304
constexpr int PP    = HH + 2;             // 50: padded spatial extent (halo of 1)
constexpr int PPSZ  = PP * PP;            // 2500 padded pixels
constexpr int MDIM  = BATCH * HWSZ;       // 9216 (GEMM M per timestep)
constexpr int NDIM  = 4 * KCH;            // 256  (GEMM N = 4K gates)
constexpr int KREC  = 9 * KCH;            // 576  (K of every WMMA GEMM: 3x3 x 64ch)

// ---------------- CDNA5 WMMA types ----------------
typedef __attribute__((ext_vector_type(16))) __bf16        v16bf;
typedef __attribute__((ext_vector_type(8)))  float         v8f;
typedef __attribute__((ext_vector_type(4)))  unsigned int  v4u;

union Frag16x32 { v4u u[2]; v16bf v; };   // 32B: one bf16 A/B fragment

__device__ __forceinline__ unsigned short f32_to_bf16(float f) {
    unsigned int u = __float_as_uint(f);
    u = (u + 0x7FFFu + ((u >> 16) & 1u)) >> 16;   // round-to-nearest-even
    return (unsigned short)u;
}
__device__ __forceinline__ float sigmoidf(float x) {
    return 1.0f / (1.0f + __expf(-x));
}

// ---------------- small helper kernels ----------------
__global__ void zero_u32(unsigned int* __restrict__ p, int n) {
    int i = blockIdx.x * blockDim.x + threadIdx.x;
    if (i < n) p[i] = 0u;
}

// w: [256][cin_tot][3][3] f32 -> WmatT: [256][576] bf16, k = (r*3+s)*64 + ci,
// taking 64 input channels starting at choff.
__global__ void prep_weightsT(const float* __restrict__ w,
                              unsigned short* __restrict__ wmatT,
                              int cin9tot, int choff) {
    int e = blockIdx.x * blockDim.x + threadIdx.x;     // e = n*576 + k
    if (e >= NDIM * KREC) return;
    int n  = e / KREC;
    int k  = e - n * KREC;
    int rs = k >> 6;                                   // 0..8
    int ci = k & 63;
    wmatT[e] = f32_to_bf16(w[(long)n * cin9tot + (ci + choff) * 9 + rs]);
}

// Layer-1 feed-forward contribution: xc[m][256] = conv3x3(x_t, w1[:,0,:,:]) + b1
__global__ void __launch_bounds__(256)
xcontrib(const float* __restrict__ x, int t,
         const float* __restrict__ w1, const float* __restrict__ b1,
         float* __restrict__ xc) {
    __shared__ float xv[9];
    const int m  = blockIdx.x;            // 0..9215
    const int n  = threadIdx.x;           // 0..255
    const int b  = m / HWSZ;
    const int hw = m - b * HWSZ;
    const int yr = hw / WW;
    const int xcn = hw - yr * WW;
    if (n < 9) {
        int r  = n / 3, s = n - r * 3;
        int yy = yr + r - 1, xx = xcn + s - 1;
        float v = 0.f;
        if ((unsigned)yy < (unsigned)HH && (unsigned)xx < (unsigned)WW)
            v = x[((long)b * TT + t) * HWSZ + yy * WW + xx];
        xv[n] = v;
    }
    __syncthreads();
    float acc = b1[n];
    const float* wp = w1 + (long)n * 585;  // channel 0 of w1, 9 taps contiguous
    #pragma unroll
    for (int i = 0; i < 9; ++i) acc += xv[i] * wp[i];
    xc[(long)m * NDIM + n] = acc;
}

// ---- implicit-GEMM 3x3 conv over a zero-padded 64-channel NHWC source (WMMA bf16) ----
// out[m][256] = im2col(src) @ WmatT^T + addend ; addend[m*astride + n] (astride 0 => bias)
// src: bf16 NHWC [b][50*50][64] zero-padded (halo=1), per-batch stride src_bs.
// 8 waves arranged 2(M)x4(N); each wave owns 64Mx64N (4x4 fragments): one 32-K chunk is
// 8 A-loads + 8 B-loads for 16 WMMAs. Padding is materialized, so every A load is an
// unconditional global_load_b128 at an unroll-time immediate offset from one base/frag.
// LDS-free, no barriers, no exec masks in the loop.
__global__ void __launch_bounds__(256)
conv64_gemm_wmma(const unsigned short* __restrict__ src, long src_bs,
                 const unsigned short* __restrict__ wmatT,
                 const float* __restrict__ addend, long astride,
                 float* __restrict__ outp) {
    const int tid  = threadIdx.x;
    const int lane = tid & 31;
    const int wv   = tid >> 5;                 // wave 0..7
    const int wm   = wv >> 2;                  // 0..1 : M position
    const int wn   = wv & 3;                   // 0..3 : N position
    const int mbase = blockIdx.x * 128 + wm * 64;
    const int nbase = wn * 64;

    // Per-M-fragment base pointer (16-bit A 16x32: lanes0-15 K{0-7,16-23},
    // lanes16-31 K{8-15,24-31}; row = base + (lane&15))
    const int akb = (lane >> 4) * 8;           // k offset within 32-chunk: 0 or 8
    const unsigned short* pbase[4];
    #pragma unroll
    for (int mi = 0; mi < 4; ++mi) {
        const int am  = mbase + mi * 16 + (lane & 15);
        const int ab  = am / HWSZ;
        const int ahw = am - ab * HWSZ;
        const int ay  = ahw / WW;
        const int ax  = ahw - ay * WW;
        // top-left tap (r=0,s=0) of this output pixel in the padded image
        pbase[mi] = src + (long)ab * src_bs + (long)(ay * PP + ax) * 64 + akb;
    }

    // B column base (B 32x16: lanes0-15 K{0-7,16-23}, lanes16-31 K{8-15,24-31}; col = lane&15)
    const unsigned short* wb = wmatT + (long)(nbase + (lane & 15)) * KREC;
    const int kbf = (lane >> 4) * 8;

    v8f acc[4][4] = {};                        // [j (N frag)][mi (M frag)]

    #pragma unroll
    for (int rs = 0; rs < 9; ++rs) {
        const int r = rs / 3;
        const int s = rs - r * 3;
        const int toff = (r * PP + s) * 64;          // unroll-time constant
        #pragma unroll
        for (int half = 0; half < 2; ++half) {       // ci0 = half*32
            Frag16x32 fa[4];
            #pragma unroll
            for (int mi = 0; mi < 4; ++mi) {
                const unsigned short* pa = pbase[mi] + toff + half * 32;
                fa[mi].u[0] = *(const v4u*)pa;           // ci [kb, kb+8)
                fa[mi].u[1] = *(const v4u*)(pa + 16);    // ci [kb+16, kb+24)
            }
            const int kk = (rs * 2 + half) * 32 + kbf;
            #pragma unroll
            for (int j = 0; j < 4; ++j) {
                const unsigned short* pb = wb + (long)j * 16 * KREC + kk;
                Frag16x32 fb;
                fb.u[0] = *(const v4u*)pb;               // K [kk, kk+8)
                fb.u[1] = *(const v4u*)(pb + 16);        // K [kk+16, kk+24)
                #pragma unroll
                for (int mi = 0; mi < 4; ++mi) {
                    acc[j][mi] = __builtin_amdgcn_wmma_f32_16x16x32_bf16(
                        false, fa[mi].v, false, fb.v, (short)0, acc[j][mi],
                        false, false);
                }
            }
        }
    }

    // D layout: VGPR r -> lanes0-15 M=r, lanes16-31 M=8+r; N = lane&15
    const int hi = lane >> 4;
    const int nl = lane & 15;
    #pragma unroll
    for (int j = 0; j < 4; ++j) {
        const int ng = nbase + j * 16 + nl;
        #pragma unroll
        for (int mi = 0; mi < 4; ++mi) {
            const int mg0 = mbase + mi * 16 + hi * 8;
            #pragma unroll
            for (int rr = 0; rr < 8; ++rr) {
                const long mg = mg0 + rr;
                outp[mg * NDIM + ng] = acc[j][mi][rr] + addend[mg * astride + ng];
            }
        }
    }
}

// ---------------- LSTM pointwise update ----------------
// gates[m][256] (i,f,g,o blocks of 64); c: [b][kc][hw];
// h -> padded NHWC bf16 [b][50*50][64] (interior only); y (f32, BN input): [b][kc][t][hw]
__global__ void __launch_bounds__(256)
lstm_update(const float* __restrict__ gates, float* __restrict__ c,
            unsigned short* __restrict__ hbf, float* __restrict__ yout, int t) {
    int e = blockIdx.x * blockDim.x + threadIdx.x;
    constexpr int TOT = BATCH * KCH * HWSZ;
    if (e >= TOT) return;
    int b   = e / (KCH * HWSZ);
    int rem = e - b * (KCH * HWSZ);
    int kc  = rem / HWSZ;
    int hw  = rem - kc * HWSZ;
    const float* g = gates + (long)(b * HWSZ + hw) * NDIM;
    float gi = sigmoidf(g[kc]);
    float gf = sigmoidf(g[KCH + kc]);
    float gg = tanhf(g[2 * KCH + kc]);
    float go = sigmoidf(g[3 * KCH + kc]);
    float cc = gf * c[e] + gi * gg;
    c[e] = cc;
    float hh = go * tanhf(cc);
    int yy = hw / WW, xx = hw - yy * WW;
    hbf[((long)b * PPSZ + (yy + 1) * PP + (xx + 1)) * 64 + kc] = f32_to_bf16(hh);
    yout[((long)(b * KCH + kc) * TT + t) * HWSZ + hw] = hh;
}

// ---------------- per-channel training-mode BatchNorm ----------------
// Input y: f32 [b][ch][t][hw].
// BF16OUT=1: write bf16 padded NHWC [b][t][50*50][64] interior (feeds next conv).
// BF16OUT=0: write f32 [b][ch][t][hw] (final output).
template <bool BF16OUT>
__global__ void __launch_bounds__(256)
bn_channel(const float* __restrict__ yin, const float* __restrict__ gamma,
           const float* __restrict__ beta, void* __restrict__ outp) {
    __shared__ float s1[256], s2[256];
    const int  ch  = blockIdx.x;
    const int  tid = threadIdx.x;
    const long thw = (long)TT * HWSZ;            // 36864
    const long NE  = (long)BATCH * thw;          // 147456
    float sum = 0.f, sq = 0.f;
    for (long e = tid; e < NE; e += 256) {
        long b   = e / thw;
        long rem = e - b * thw;
        float v  = yin[(b * KCH + ch) * thw + rem];
        sum += v; sq += v * v;
    }
    s1[tid] = sum; s2[tid] = sq;
    __syncthreads();
    for (int s = 128; s > 0; s >>= 1) {
        if (tid < s) { s1[tid] += s1[tid + s]; s2[tid] += s2[tid + s]; }
        __syncthreads();
    }
    const float mean = s1[0] / (float)NE;
    const float var  = s2[0] / (float)NE - mean * mean;
    const float scl  = rsqrtf(var + 1e-5f) * gamma[ch];
    const float off  = beta[ch] - mean * scl;
    for (long e = tid; e < NE; e += 256) {
        long b   = e / thw;
        long rem = e - b * thw;
        float v  = yin[(b * KCH + ch) * thw + rem] * scl + off;
        if (BF16OUT) {
            long t  = rem / HWSZ;
            long hw = rem - t * HWSZ;
            long yy = hw / WW, xx = hw - yy * WW;
            ((unsigned short*)outp)[(((long)b * TT + t) * PPSZ +
                                     (yy + 1) * PP + (xx + 1)) * 64 + ch] =
                f32_to_bf16(v);
        } else {
            ((float*)outp)[(b * KCH + ch) * thw + rem] = v;
        }
    }
}

// ---------------- host orchestration ----------------
extern "C" void kernel_launch(void* const* d_in, const int* in_sizes, int n_in,
                              void* d_out, int out_size, void* d_ws, size_t ws_size,
                              hipStream_t stream) {
    (void)in_sizes; (void)n_in; (void)out_size; (void)ws_size;
    const float* x      = (const float*)d_in[0];
    const float* w1     = (const float*)d_in[1];
    const float* b1     = (const float*)d_in[2];
    const float* gamma1 = (const float*)d_in[3];
    const float* beta1  = (const float*)d_in[4];
    const float* w2     = (const float*)d_in[5];
    const float* b2     = (const float*)d_in[6];
    const float* gamma2 = (const float*)d_in[7];
    const float* beta2  = (const float*)d_in[8];
    float* out = (float*)d_out;

    // workspace carve-out (256B aligned; hbf and cst contiguous for one-shot zero)
    char* p = (char*)d_ws;
    auto carve = [&](size_t bytes) {
        void* r = (void*)p;
        p += (bytes + 255) & ~(size_t)255;
        return r;
    };
    unsigned short* wh1T = (unsigned short*)carve((size_t)NDIM * KREC * 2);  // 294912
    unsigned short* wy2T = (unsigned short*)carve((size_t)NDIM * KREC * 2);
    unsigned short* wh2T = (unsigned short*)carve((size_t)NDIM * KREC * 2);
    unsigned short* hbf  = (unsigned short*)carve((size_t)BATCH * PPSZ * 64 * 2);   // 1.28MB padded
    float*          cst  = (float*)carve((size_t)BATCH * KCH * HWSZ * 4);           // 2.36MB
    float*          gts  = (float*)carve((size_t)MDIM * NDIM * 4);                  // 9.4MB
    float*          ffc  = (float*)carve((size_t)MDIM * NDIM * 4);                  // 9.4MB (xc / yc)
    float*          ybuf = (float*)carve((size_t)BATCH * KCH * TT * HWSZ * 4);      // 37.7MB
    unsigned short* y1bn = (unsigned short*)carve((size_t)BATCH * TT * PPSZ * 64 * 2); // 20.5MB padded

    const int  nhc   = (BATCH * PPSZ * 64 * 2 + BATCH * KCH * HWSZ * 4) / 4;   // 909824 u32
    const int  ny1   = (BATCH * TT * PPSZ * 64 * 2) / 4;                       // 5120000 u32
    const dim3 gGemm(MDIM / 128, 1, 1);                                        // 72 blocks
    const int  nLstmBlk = (BATCH * KCH * HWSZ + 255) / 256;                    // 2304
    const int  nPrepBlk = (NDIM * KREC + 255) / 256;                           // 576

    // one-time: weights + zero the padded y1bn staging (borders stay zero forever)
    prep_weightsT<<<nPrepBlk, 256, 0, stream>>>(w1, wh1T, 65 * 9, 1);
    prep_weightsT<<<nPrepBlk, 256, 0, stream>>>(w2, wy2T, 128 * 9, 0);
    prep_weightsT<<<nPrepBlk, 256, 0, stream>>>(w2, wh2T, 128 * 9, 64);
    zero_u32<<<(ny1 + 255) / 256, 256, 0, stream>>>((unsigned int*)y1bn, ny1);

    // -------- layer 1 --------
    zero_u32<<<(nhc + 255) / 256, 256, 0, stream>>>((unsigned int*)hbf, nhc);
    for (int t = 0; t < TT; ++t) {
        xcontrib<<<MDIM, 256, 0, stream>>>(x, t, w1, b1, ffc);
        conv64_gemm_wmma<<<gGemm, 256, 0, stream>>>(
            hbf, (long)PPSZ * 64, wh1T, ffc, (long)NDIM, gts);
        lstm_update<<<nLstmBlk, 256, 0, stream>>>(gts, cst, hbf, ybuf, t);
    }
    bn_channel<true><<<KCH, 256, 0, stream>>>(ybuf, gamma1, beta1, (void*)y1bn);

    // -------- layer 2 --------
    zero_u32<<<(nhc + 255) / 256, 256, 0, stream>>>((unsigned int*)hbf, nhc);
    for (int t = 0; t < TT; ++t) {
        // feed-forward contribution from BN'd layer-1 output (bias folded in)
        conv64_gemm_wmma<<<gGemm, 256, 0, stream>>>(
            y1bn + (long)t * PPSZ * 64, (long)TT * PPSZ * 64, wy2T, b2, 0L, ffc);
        // recurrent contribution from h2, chained on top
        conv64_gemm_wmma<<<gGemm, 256, 0, stream>>>(
            hbf, (long)PPSZ * 64, wh2T, ffc, (long)NDIM, gts);
        lstm_update<<<nLstmBlk, 256, 0, stream>>>(gts, cst, hbf, ybuf, t);
    }
    bn_channel<false><<<KCH, 256, 0, stream>>>(ybuf, gamma2, beta2, (void*)out);
}
